// PhiMoESparseMoeBlock_33062658245117
// MI455X (gfx1250) — matrix operand order repaired and checked by reading, not compiled, not run on hardware
//
#include <hip/hip_runtime.h>
#include <hip/hip_bf16.h>
#include <math.h>

#define E 8
#define H 1024
#define F 3584
#define T 2048
#define JE 0.01f
#define MT 16     // tokens per block tile
#define FC 128    // F-chunk width
#define XPAD 8
#define APAD 8

typedef __attribute__((ext_vector_type(16))) __bf16 v16bf;
typedef __attribute__((ext_vector_type(8)))  float  v8f;

union Frag {
    v16bf v;
    unsigned short u[16];
    unsigned int   w[8];
    uint4 q[2];
};

__device__ __forceinline__ unsigned short f2bf(float f) {
    unsigned int x = __float_as_uint(f);
    x += 0x7FFFu + ((x >> 16) & 1u);   // round-to-nearest-even
    return (unsigned short)(x >> 16);
}

// packed fp32x2 -> bf16x2 (v_cvt_pk_bf16_f32 when available)
__device__ __forceinline__ unsigned int pk_bf16(float a, float b) {
#if __has_builtin(__builtin_amdgcn_cvt_pk_bf16_f32)
    auto r = __builtin_amdgcn_cvt_pk_bf16_f32(a, b);
    unsigned int u;
    __builtin_memcpy(&u, &r, 4);
    return u;
#else
    return (unsigned int)f2bf(a) | ((unsigned int)f2bf(b) << 16);
#endif
}

// ---- B-operand loaders: lane holds one N-column, 16 consecutive K values ----
template <typename WT>
__device__ __forceinline__ void loadBW(const WT* __restrict__ p, Frag& b);

template <>
__device__ __forceinline__ void loadBW<float>(const float* __restrict__ p, Frag& b) {
    float4 f0 = *(const float4*)(p);
    float4 f1 = *(const float4*)(p + 4);
    float4 f2 = *(const float4*)(p + 8);
    float4 f3 = *(const float4*)(p + 12);
    b.w[0] = pk_bf16(f0.x, f0.y); b.w[1] = pk_bf16(f0.z, f0.w);
    b.w[2] = pk_bf16(f1.x, f1.y); b.w[3] = pk_bf16(f1.z, f1.w);
    b.w[4] = pk_bf16(f2.x, f2.y); b.w[5] = pk_bf16(f2.z, f2.w);
    b.w[6] = pk_bf16(f3.x, f3.y); b.w[7] = pk_bf16(f3.z, f3.w);
}

template <>
__device__ __forceinline__ void loadBW<unsigned short>(const unsigned short* __restrict__ p, Frag& b) {
    b.q[0] = *(const uint4*)(p);       // 8 bf16
    b.q[1] = *(const uint4*)(p + 8);   // 8 bf16
}

// ---------------- kernel 0: zero the output (we accumulate with atomics) ----------------
__global__ void zero_out_kernel(float* __restrict__ out, int n4) {
    int i = blockIdx.x * blockDim.x + threadIdx.x;
    if (i < n4) ((float4*)out)[i] = make_float4(0.f, 0.f, 0.f, 0.f);
}

// ---------------- kernel 0b: fp32 -> bf16 weight pre-conversion ----------------
__global__ __launch_bounds__(256) void cvt_bf16_kernel(const float* __restrict__ src,
                                                       unsigned short* __restrict__ dst, int n8) {
    int i = blockIdx.x * blockDim.x + threadIdx.x;   // 8 elements per thread
    if (i < n8) {
        const float4* s = (const float4*)src;
        float4 a = s[2 * i];
        float4 b = s[2 * i + 1];
        uint4 o;
        o.x = pk_bf16(a.x, a.y); o.y = pk_bf16(a.z, a.w);
        o.z = pk_bf16(b.x, b.y); o.w = pk_bf16(b.z, b.w);
        ((uint4*)dst)[i] = o;
    }
}

// ---------------- kernel 1: routing (scores + jitter-masked top-2 softmax) ----------------
__global__ __launch_bounds__(256) void route_kernel(const float* __restrict__ x,
                                                    const float* __restrict__ gw,
                                                    float* __restrict__ g) {
    const int t   = blockIdx.x;
    const int tid = threadIdx.x;
    __shared__ float sred[E][256];

    float part[E];
#pragma unroll
    for (int e = 0; e < E; ++e) part[e] = 0.f;
    const float* xr = x + (size_t)t * H;
    for (int h = tid; h < H; h += 256) {
        float xv = xr[h];
#pragma unroll
        for (int e = 0; e < E; ++e) part[e] += xv * gw[e * H + h];
    }
#pragma unroll
    for (int e = 0; e < E; ++e) sred[e][tid] = part[e];
    __syncthreads();
    for (int s = 128; s > 0; s >>= 1) {
        if (tid < s) {
#pragma unroll
            for (int e = 0; e < E; ++e) sred[e][tid] += sred[e][tid + s];
        }
        __syncthreads();
    }

    if (tid == 0) {
        float s[E];
#pragma unroll
        for (int e = 0; e < E; ++e) s[e] = sred[e][0];
        int i0 = 0;
#pragma unroll
        for (int e = 1; e < E; ++e) if (s[e] > s[i0]) i0 = e;
        float t1 = s[i0];
        int i1 = (i0 == 0) ? 1 : 0;
#pragma unroll
        for (int e = 0; e < E; ++e) if (e != i0 && s[e] > s[i1]) i1 = e;
        float t2 = s[i1];

        float sum1 = 0.f, sum2 = 0.f;
#pragma unroll
        for (int e = 0; e < E; ++e) {
            float fac1 = fmaxf(fabsf(s[e]), t1);
            bool keep1 = !(((t1 - s[e]) / fac1) > 2.f * JE);
            if (keep1) sum1 += __expf(s[e] - t1);
            float fac2 = fmaxf(fabsf(s[e]), t2);
            bool keep2 = (e != i0) && !(((t2 - s[e]) / fac2) > 2.f * JE);
            if (keep2) sum2 += __expf(s[e] - t2);
        }
        float m1 = 1.f / sum1;
        float m2 = 1.f / sum2;
#pragma unroll
        for (int e = 0; e < E; ++e)
            g[(size_t)t * E + e] = (e == i0) ? m1 : ((e == i1) ? m2 : 0.f);
    }
}

// ---------------- kernel 2: deterministic per-expert token gather (1 wave / expert) --------
__global__ void gather_kernel(const float* __restrict__ g, int* __restrict__ counts,
                              int* __restrict__ lists) {
    const int e    = blockIdx.x;
    const int lane = threadIdx.x;   // one wave32
    int cnt = 0;
    for (int base = 0; base < T; base += 32) {
        int t = base + lane;
        bool p = g[(size_t)t * E + e] != 0.f;
        unsigned long long m = __ballot(p);
        int pre = __popcll(m & ((1ull << lane) - 1ull));
        if (p) lists[e * T + cnt + pre] = t;
        cnt += __popcll(m);
    }
    if (lane == 0) counts[e] = cnt;
}

// ---------------- kernel 3: fused expert MLP (WMMA bf16) ----------------
// grid: (T/MT tiles, E experts), block: 256 threads = 8 waves
template <typename WT>
__global__ __launch_bounds__(256) void moe_kernel(
        const float* __restrict__ x,  const WT* __restrict__ w1,
        const WT* __restrict__ w2,    const WT* __restrict__ w3,
        const float* __restrict__ g,  const int* __restrict__ counts,
        const int* __restrict__ lists, float* __restrict__ out) {
    const int e    = blockIdx.y;
    const int tile = blockIdx.x;
    const int count = counts[e];
    if (tile * MT >= count) return;

    __shared__ unsigned short xA[MT][H + XPAD];    // x tile, bf16, row-major
    __shared__ unsigned short actA[MT][FC + APAD]; // silu(h1)*h3*g chunk, bf16
    __shared__ float gv[MT];
    __shared__ int   toks[MT];

    const int tid  = threadIdx.x;
    const int lane = tid & 31;
    const int wave = tid >> 5;     // 0..7
    const int nlo  = lane & 15;    // row (A) / column (B,C,D) index
    const int hi   = lane >> 4;    // lane half selects K sub-range

    if (tid < MT) {
        int slot = tile * MT + tid;
        int tok; float gg;
        if (slot < count) {
            tok = lists[e * T + slot];
            gg  = g[(size_t)tok * E + e];
        } else {
            tok = lists[e * T];   // valid address, zero contribution
            gg  = 0.f;
        }
        toks[tid] = tok;
        gv[tid]   = gg;
    }
    __syncthreads();

    // stage x rows fp32 -> bf16 in LDS (thread = 1/16 of a row)
    {
        int r  = tid >> 4;
        int c0 = (tid & 15) * 64;
        const float* xr = x + (size_t)toks[r] * H + c0;
#pragma unroll
        for (int c = 0; c < 64; c += 8) {
            float4 a = *(const float4*)(xr + c);
            float4 b = *(const float4*)(xr + c + 4);
            uint4 o;
            o.x = pk_bf16(a.x, a.y); o.y = pk_bf16(a.z, a.w);
            o.z = pk_bf16(b.x, b.y); o.w = pk_bf16(b.z, b.w);
            *(uint4*)&xA[r][c0 + c] = o;
        }
    }
    __syncthreads();

    v8f yacc[8];
#pragma unroll
    for (int i = 0; i < 8; ++i) yacc[i] = (v8f){};

    for (int fc = 0; fc < F; fc += FC) {
        // ---- Phase A: h1/h3 tile (16 tokens x 16 f-cols per wave), K over H ----
        v8f acc1 = {};
        v8f acc3 = {};
        const int fcol = fc + wave * 16 + nlo;
        const WT* w1row = w1 + ((size_t)e * F + fcol) * H + hi * 16;
        const WT* w3row = w3 + ((size_t)e * F + fcol) * H + hi * 16;
#pragma unroll 1
        for (int h = 0; h < H; h += 32) {
            Frag a, b1, b3;
            const unsigned short* xr = &xA[nlo][h + hi * 16];
            a.q[0] = *(const uint4*)(xr);
            a.q[1] = *(const uint4*)(xr + 8);
            loadBW<WT>(w1row + h, b1);
            loadBW<WT>(w3row + h, b3);
            acc1 = __builtin_amdgcn_wmma_f32_16x16x32_bf16(false, a.v, false, b1.v,
                                                           (short)0, acc1, false, false);
            acc3 = __builtin_amdgcn_wmma_f32_16x16x32_bf16(false, a.v, false, b3.v,
                                                           (short)0, acc3, false, false);
        }
        // act = silu(h1) * h3 * g  (cheap sigmoid: v_rcp_f32, not IEEE divide)
#pragma unroll
        for (int r = 0; r < 8; ++r) {
            int m = r + hi * 8;
            float h1v = acc1[r];
            float h3v = acc3[r];
            float sg  = __builtin_amdgcn_rcpf(1.f + __expf(-h1v));
            float av  = h1v * sg * h3v * gv[m];
            actA[m][wave * 16 + nlo] = f2bf(av);
        }
        __syncthreads();

        // ---- Phase B: y += act @ w2_chunk^T (wave owns 128 H-columns) ----
#pragma unroll 1
        for (int ks = 0; ks < FC; ks += 32) {
            Frag a;
            const unsigned short* ar = &actA[nlo][ks + hi * 16];
            a.q[0] = *(const uint4*)(ar);
            a.q[1] = *(const uint4*)(ar + 8);
#pragma unroll
            for (int nt = 0; nt < 8; ++nt) {
                int hcol = wave * 128 + nt * 16 + nlo;
                const WT* w2row = w2 + ((size_t)e * H + hcol) * F + fc + ks + hi * 16;
                Frag b2;
                loadBW<WT>(w2row, b2);
                yacc[nt] = __builtin_amdgcn_wmma_f32_16x16x32_bf16(false, a.v, false, b2.v,
                                                                   (short)0, yacc[nt], false, false);
            }
        }
        __syncthreads();
    }

    // ---- write-out: each token appears in exactly 2 experts -> atomic add ----
#pragma unroll
    for (int nt = 0; nt < 8; ++nt) {
        int hcol = wave * 128 + nt * 16 + nlo;
#pragma unroll
        for (int r = 0; r < 8; ++r) {
            int m = r + hi * 8;
            unsafeAtomicAdd(&out[(size_t)toks[m] * H + hcol], yacc[nt][r]);
        }
    }
}

// ---------------- host launcher ----------------
extern "C" void kernel_launch(void* const* d_in, const int* in_sizes, int n_in,
                              void* d_out, int out_size, void* d_ws, size_t ws_size,
                              hipStream_t stream) {
    const float* x  = (const float*)d_in[0];   // (2,1024,H) fp32
    const float* gw = (const float*)d_in[1];   // (E,H)
    const float* w1 = (const float*)d_in[2];   // (E,F,H)
    const float* w2 = (const float*)d_in[3];   // (E,H,F)
    const float* w3 = (const float*)d_in[4];   // (E,F,H)
    float* out = (float*)d_out;                // (2,1024,H) fp32

    char*  ws     = (char*)d_ws;
    float* g      = (float*)ws;                                       // T*E floats
    int*   counts = (int*)(ws + (size_t)T * E * sizeof(float));       // E ints
    int*   lists  = (int*)(ws + (size_t)T * E * sizeof(float) + 256); // E*T ints

    const size_t wboff = (size_t)T * E * 4 + 256 + (size_t)E * T * 4; // 131328 (256-aligned)
    const size_t wsz   = (size_t)E * F * H;                           // elems per weight array
    const bool   pre   = ws_size >= wboff + 3 * wsz * sizeof(unsigned short);

    int n4 = out_size / 4;
    zero_out_kernel<<<(n4 + 255) / 256, 256, 0, stream>>>(out, n4);
    route_kernel<<<T, 256, 0, stream>>>(x, gw, g);
    gather_kernel<<<E, 32, 0, stream>>>(g, counts, lists);

    dim3 grid(T / MT, E);
    if (pre) {
        unsigned short* w1b = (unsigned short*)(ws + wboff);
        unsigned short* w3b = w1b + wsz;
        unsigned short* w2b = w3b + wsz;
        int n8 = (int)(wsz / 8);
        int nb = (n8 + 255) / 256;
        cvt_bf16_kernel<<<nb, 256, 0, stream>>>(w1, w1b, n8);
        cvt_bf16_kernel<<<nb, 256, 0, stream>>>(w3, w3b, n8);
        cvt_bf16_kernel<<<nb, 256, 0, stream>>>(w2, w2b, n8);
        moe_kernel<unsigned short><<<grid, 256, 0, stream>>>(x, w1b, w2b, w3b,
                                                             g, counts, lists, out);
    } else {
        moe_kernel<float><<<grid, 256, 0, stream>>>(x, w1, w2, w3,
                                                    g, counts, lists, out);
    }
}
